// MoE_40501541601518
// MI455X (gfx1250) — matrix operand order repaired and checked by reading, not compiled, not run on hardware
//
#include <hip/hip_runtime.h>
#include <hip/hip_bf16.h>

#define TOKENS 8192
#define DIN    1024
#define DOUT   1024
#define NEXP   8

// GEMM tiling: block 128x128, K-chunk 32, 8 waves (2x4) of 64x32 (8 accumulators)
#define BM 128
#define BN 128
#define BK 32
#define LDT 40              // LDS row stride (bf16 elems): 80B rows, 16B aligned
#define NCH (DIN / BK)      // 32 K-chunks

typedef __attribute__((ext_vector_type(16))) __bf16        v16bf;
typedef __attribute__((ext_vector_type(8)))  float         v8f;
typedef __attribute__((ext_vector_type(4)))  unsigned int  u32x4;

union FragBF {
    v16bf bf;
    u32x4 q[2];
};

__device__ __forceinline__ unsigned short f2bf(float f) {
    unsigned int u = __float_as_uint(f);
    unsigned int r = u + 0x7FFFu + ((u >> 16) & 1u);   // round-to-nearest-even
    return (unsigned short)(r >> 16);
}

// ---------------------------------------------------------------------------
// Kernel 1: convert expert weights fp32 -> bf16
// ---------------------------------------------------------------------------
__global__ void cvt_w_kernel(const float* __restrict__ w,
                             unsigned short* __restrict__ wb, int n4) {
    int i = blockIdx.x * blockDim.x + threadIdx.x;   // 4 elements per thread
    if (i >= n4) return;
    float4 f = ((const float4*)w)[i];
    ushort4 o;
    o.x = f2bf(f.x); o.y = f2bf(f.y); o.z = f2bf(f.z); o.w = f2bf(f.w);
    ((ushort4*)wb)[i] = o;
}

// ---------------------------------------------------------------------------
// Kernel 2: router (one wave32 per token) + x fp32->bf16 conversion
// ---------------------------------------------------------------------------
__global__ void router_kernel(const float* __restrict__ x,
                              const float* __restrict__ wr,
                              const float* __restrict__ br,
                              unsigned short* __restrict__ xb,
                              int* __restrict__ ids,
                              int* __restrict__ lists,
                              int* __restrict__ counts) {
    const int lane  = threadIdx.x & 31;
    const int wave  = threadIdx.x >> 5;
    const int token = blockIdx.x * 8 + wave;         // grid = TOKENS/8 blocks of 256

    float acc[NEXP];
#pragma unroll
    for (int e = 0; e < NEXP; ++e) acc[e] = 0.0f;

    const float*    xrow  = x  + (size_t)token * DIN;
    unsigned short* xbrow = xb + (size_t)token * DIN;

    for (int i = 0; i < DIN / 32; ++i) {
        int d = i * 32 + lane;
        float xv = xrow[d];
        xbrow[d] = f2bf(xv);
#pragma unroll
        for (int e = 0; e < NEXP; ++e) acc[e] += xv * wr[e * DIN + d];
    }
#pragma unroll
    for (int e = 0; e < NEXP; ++e) {
#pragma unroll
        for (int off = 16; off >= 1; off >>= 1)
            acc[e] += __shfl_xor(acc[e], off, 32);
    }
    if (lane == 0) {
        float z[NEXP];
#pragma unroll
        for (int e = 0; e < NEXP; ++e) z[e] = acc[e] + br[e];
        // top-2, ties -> lower index (strict >), matches jax.lax.top_k
        int e0 = 0; float m0v = z[0];
#pragma unroll
        for (int e = 1; e < NEXP; ++e) if (z[e] > m0v) { m0v = z[e]; e0 = e; }
        int e1 = 0; float m1v = -3.4e38f;
#pragma unroll
        for (int e = 0; e < NEXP; ++e) if (e != e0 && z[e] > m1v) { m1v = z[e]; e1 = e; }
        ids[2 * token]     = e0;
        ids[2 * token + 1] = e1;
        int p0 = atomicAdd(&counts[e0], 1);
        lists[e0 * TOKENS + p0] = 2 * token;
        int p1 = atomicAdd(&counts[e1], 1);
        lists[e1 * TOKENS + p1] = 2 * token + 1;
    }
}

// ---------------------------------------------------------------------------
// Kernel 3: grouped expert GEMM, bf16 WMMA, double-buffered LDS tiles.
// grid = (DOUT/BN, TOKENS/BM, NEXP); block = 256 (8 waves, 2 row-bands x 4
// col-bands; each wave computes a 64x32 tile = 8 independent 16x16 wmma accs)
// ---------------------------------------------------------------------------
__global__ void __launch_bounds__(256)
moe_gemm_kernel(const unsigned short* __restrict__ xb,
                const unsigned short* __restrict__ wb,
                const int* __restrict__ lists,
                const int* __restrict__ counts,
                float* __restrict__ partial) {
    const int e   = blockIdx.z;
    const int cnt = counts[e];
    const int m0  = blockIdx.y * BM;
    if (m0 >= cnt) return;                            // uniform early exit
    const int n0  = blockIdx.x * BN;

    __shared__ unsigned short sA[2][BM * LDT];        // 2 x 10240 B
    __shared__ unsigned short sB[2][BN * LDT];        // 2 x 10240 B
    __shared__ int sRow[BM];
    __shared__ int sTok[BM];

    const int tid = threadIdx.x;
    if (tid < BM) {
        int idx = m0 + tid;
        int v = lists[e * TOKENS + (idx < cnt ? idx : cnt - 1)];
        sRow[tid] = v;
        sTok[tid] = v >> 1;                           // token index
    }
    __syncthreads();

    // cooperative global->LDS coordinates (A and B tiles are both 128x32)
    const int ar = tid >> 1;                          // 0..127 : tile row
    const int ak = (tid & 1) * 16;                    // 16 elems (2 x b128) each
    const unsigned short* aSrc = xb + (size_t)sTok[ar] * DIN + ak;
    const unsigned short* bSrc = wb + ((size_t)e * DOUT + n0 + ar) * DIN + ak;

    const int lane = tid & 31;
    const int wv   = tid >> 5;
    const int wm   = wv >> 2;        // 0..1  : 64-row band
    const int wn   = wv & 3;         // 0..3  : 32-col band
    const int l15  = lane & 15;
    const int kq   = (lane >> 4) * 8;     // A frag chunk base (ISA layout)
    const int kb   = (lane >> 4) * 8;     // B frag chunk base (ISA layout)

    v8f acc[4][2];
#pragma unroll
    for (int i = 0; i < 4; ++i)
#pragma unroll
        for (int j = 0; j < 2; ++j)
            acc[i][j] = (v8f){0.f,0.f,0.f,0.f,0.f,0.f,0.f,0.f};

    u32x4 ra0, ra1, rb0, rb1;

    // -------- prologue: stage chunk 0 --------
    ra0 = *(const u32x4*)(aSrc);
    ra1 = *(const u32x4*)(aSrc + 8);
    rb0 = *(const u32x4*)(bSrc);
    rb1 = *(const u32x4*)(bSrc + 8);
    *(u32x4*)&sA[0][ar * LDT + ak]     = ra0;
    *(u32x4*)&sA[0][ar * LDT + ak + 8] = ra1;
    *(u32x4*)&sB[0][ar * LDT + ak]     = rb0;
    *(u32x4*)&sB[0][ar * LDT + ak + 8] = rb1;
    __syncthreads();

    for (int ch = 0; ch < NCH; ++ch) {
        const int cur = ch & 1;
        const int kn  = (ch + 1) * BK;
        if (ch + 1 < NCH) {               // issue next chunk's global loads early
            ra0 = *(const u32x4*)(aSrc + kn);
            ra1 = *(const u32x4*)(aSrc + kn + 8);
            rb0 = *(const u32x4*)(bSrc + kn);
            rb1 = *(const u32x4*)(bSrc + kn + 8);
        }

        // fragments: 4 x A (64 rows), 2 x B (32 cols)  ->  8 wmma
        FragBF a[4], b[2];
        const unsigned short* pa = &sA[cur][(wm * 64 + l15) * LDT];
#pragma unroll
        for (int mt = 0; mt < 4; ++mt) {
            a[mt].q[0] = *(const u32x4*)(pa + mt * 16 * LDT + kq);
            a[mt].q[1] = *(const u32x4*)(pa + mt * 16 * LDT + kq + 16);
        }
        const unsigned short* pb = &sB[cur][(wn * 32 + l15) * LDT];
#pragma unroll
        for (int nt = 0; nt < 2; ++nt) {
            b[nt].q[0] = *(const u32x4*)(pb + nt * 16 * LDT + kb);
            b[nt].q[1] = *(const u32x4*)(pb + nt * 16 * LDT + kb + 16);
        }
#pragma unroll
        for (int mt = 0; mt < 4; ++mt)
#pragma unroll
            for (int nt = 0; nt < 2; ++nt)
                acc[mt][nt] = __builtin_amdgcn_wmma_f32_16x16x32_bf16(
                    false, a[mt].bf, false, b[nt].bf, (short)0, acc[mt][nt],
                    false, false);

        if (ch + 1 < NCH) {               // fill the other buffer
            const int nb = cur ^ 1;
            *(u32x4*)&sA[nb][ar * LDT + ak]     = ra0;
            *(u32x4*)&sA[nb][ar * LDT + ak + 8] = ra1;
            *(u32x4*)&sB[nb][ar * LDT + ak]     = rb0;
            *(u32x4*)&sB[nb][ar * LDT + ak + 8] = rb1;
        }
        __syncthreads();
    }

    // -------- epilogue: scatter 8 C tiles to partial rows (no atomics) --------
    const int colg = n0 + wn * 32 + l15;
    const int mb   = wm * 64 + 8 * (lane >> 4);
#pragma unroll
    for (int mt = 0; mt < 4; ++mt) {
#pragma unroll
        for (int r = 0; r < 8; ++r) {
            int ml = mb + mt * 16 + r;
            if (m0 + ml < cnt) {
                size_t row = (size_t)sRow[ml] * DOUT;
                partial[row + colg]      = acc[mt][0][r];
                partial[row + colg + 16] = acc[mt][1][r];
            }
        }
    }
}

// ---------------------------------------------------------------------------
// Kernel 4: y = partial[2t] + partial[2t+1] + b[e0] + b[e1]
// ---------------------------------------------------------------------------
__global__ void combine_kernel(const float* __restrict__ partial,
                               const int* __restrict__ ids,
                               const float* __restrict__ be,
                               float* __restrict__ y) {
    const int token = blockIdx.x;
    const int o = threadIdx.x * 4;
    const int e0 = ids[2 * token], e1 = ids[2 * token + 1];
    float4 p0 = *(const float4*)&partial[(size_t)(2 * token)     * DOUT + o];
    float4 p1 = *(const float4*)&partial[(size_t)(2 * token + 1) * DOUT + o];
    float4 b0 = *(const float4*)&be[(size_t)e0 * DOUT + o];
    float4 b1 = *(const float4*)&be[(size_t)e1 * DOUT + o];
    float4 r;
    r.x = p0.x + p1.x + b0.x + b1.x;
    r.y = p0.y + p1.y + b0.y + b1.y;
    r.z = p0.z + p1.z + b0.z + b1.z;
    r.w = p0.w + p1.w + b0.w + b1.w;
    *(float4*)&y[(size_t)token * DOUT + o] = r;
}

// ---------------------------------------------------------------------------
extern "C" void kernel_launch(void* const* d_in, const int* in_sizes, int n_in,
                              void* d_out, int out_size, void* d_ws, size_t ws_size,
                              hipStream_t stream) {
    const float* x  = (const float*)d_in[0];   // [4,2048,1024]
    const float* wr = (const float*)d_in[1];   // [8,1024]
    const float* br = (const float*)d_in[2];   // [8]
    const float* we = (const float*)d_in[3];   // [8,1024,1024]
    const float* be = (const float*)d_in[4];   // [8,1024]
    float* y = (float*)d_out;                  // [4,2048,1024]

    char* ws = (char*)d_ws;
    int*            counts  = (int*)ws;                                   // 256 B
    int*            ids     = (int*)(ws + 256);                           // 64 KB
    int*            lists   = (int*)(ws + 256 + 65536);                   // 256 KB
    unsigned short* xb      = (unsigned short*)(ws + 256 + 65536 + 262144);
    unsigned short* wbf     = xb + (size_t)TOKENS * DIN;                  // 16 MB each
    float*          partial = (float*)((char*)(wbf + (size_t)NEXP * DOUT * DIN));

    hipMemsetAsync(counts, 0, 256, stream);

    const int n4 = NEXP * DOUT * DIN / 4;
    cvt_w_kernel<<<(n4 + 255) / 256, 256, 0, stream>>>(we, wbf, n4);

    router_kernel<<<TOKENS / 8, 256, 0, stream>>>(x, wr, br, xb, ids, lists, counts);

    dim3 ggrid(DOUT / BN, TOKENS / BM, NEXP);
    moe_gemm_kernel<<<ggrid, 256, 0, stream>>>(xb, wbf, lists, counts, partial);

    combine_kernel<<<TOKENS, 256, 0, stream>>>(partial, ids, be, y);

    (void)in_sizes; (void)n_in; (void)out_size; (void)ws_size;
}
